// SparseMoE_695784702457
// MI455X (gfx1250) — matrix operand order repaired and checked by reading, not compiled, not run on hardware
//
#include <hip/hip_runtime.h>
#include <hip/hip_bf16.h>
#include <math.h>

#define D_MODEL 1024
#define N_EXPERTS 8
#define FFN_H 2048
#define LDA 1040  // 1024 + 16 bf16 pad -> 2080B row stride, 16B aligned

typedef __bf16 bf16_t;
typedef __attribute__((ext_vector_type(16))) __bf16 v16bf;
typedef __attribute__((ext_vector_type(8)))  float  v8f;

// ---------------- header layout in ws (ints) -----------------
// [0..7] counts  [8..15] fill  [16..24] offsets (exclusive scan, offsets[8]=total)

__global__ __launch_bounds__(64) void zero_header_kernel(int* hdr) {
  if (threadIdx.x < 32) hdr[threadIdx.x] = 0;
}

// ------------- router: logits, softmax, top-2, counts --------
__global__ __launch_bounds__(256) void router_kernel(
    const float* __restrict__ x, const float* __restrict__ rw, int N,
    int* __restrict__ counts, int* __restrict__ topk_idx,
    float* __restrict__ topk_w, float* __restrict__ probs)
{
  int wave = threadIdx.x >> 5, lane = threadIdx.x & 31;
  int t = blockIdx.x * 8 + wave;
  if (t >= N) return;
  const float* xr = x + (size_t)t * D_MODEL;
  float acc[N_EXPERTS];
#pragma unroll
  for (int e = 0; e < N_EXPERTS; ++e) acc[e] = 0.f;
  for (int k = lane * 4; k < D_MODEL; k += 128) {
    float4 xv = *(const float4*)(xr + k);
#pragma unroll
    for (int e = 0; e < N_EXPERTS; ++e) {
      float4 wv = *(const float4*)(rw + e * D_MODEL + k);
      acc[e] += xv.x * wv.x + xv.y * wv.y + xv.z * wv.z + xv.w * wv.w;
    }
  }
#pragma unroll
  for (int e = 0; e < N_EXPERTS; ++e) {
    float s = acc[e];
#pragma unroll
    for (int off = 16; off >= 1; off >>= 1) s += __shfl_xor(s, off, 32);
    acc[e] = s;
  }
  if (lane == 0) {
    float mx = acc[0];
#pragma unroll
    for (int e = 1; e < N_EXPERTS; ++e) mx = fmaxf(mx, acc[e]);
    float ex[N_EXPERTS], Z = 0.f;
#pragma unroll
    for (int e = 0; e < N_EXPERTS; ++e) { ex[e] = __expf(acc[e] - mx); Z += ex[e]; }
    float invZ = 1.f / Z;
    float p[N_EXPERTS];
#pragma unroll
    for (int e = 0; e < N_EXPERTS; ++e) { p[e] = ex[e] * invZ; probs[t * N_EXPERTS + e] = p[e]; }
    int i1 = 0;
#pragma unroll
    for (int e = 1; e < N_EXPERTS; ++e) if (acc[e] > acc[i1]) i1 = e;
    int i2 = (i1 == 0) ? 1 : 0;
#pragma unroll
    for (int e = 0; e < N_EXPERTS; ++e) if (e != i1 && acc[e] > acc[i2]) i2 = e;
    float wA = p[i1], wB = p[i2], s = wA + wB;
    topk_idx[2 * t] = i1;  topk_idx[2 * t + 1] = i2;
    topk_w[2 * t] = wA / s; topk_w[2 * t + 1] = wB / s;
    atomicAdd(&counts[i1], 1);
    atomicAdd(&counts[i2], 1);
  }
}

__global__ void scan_kernel(int* hdr) {
  if (threadIdx.x == 0) {
    int s = 0;
    for (int e = 0; e < N_EXPERTS; ++e) { hdr[16 + e] = s; s += hdr[e]; hdr[8 + e] = 0; }
    hdr[24] = s;
  }
}

__global__ __launch_bounds__(256) void scatter_kernel(
    const int* __restrict__ topk_idx, int* hdr,
    int* __restrict__ perm, int* __restrict__ assign_of, int twoN)
{
  int idx = blockIdx.x * 256 + threadIdx.x;
  if (idx >= twoN) return;
  int e = topk_idx[idx];
  int pos = atomicAdd(&hdr[8 + e], 1);
  int row = hdr[16 + e] + pos;
  perm[row] = idx >> 1;
  assign_of[idx] = row;
}

// map linear row-tile -> (expert, tile-in-segment); returns false if no work
__device__ __forceinline__ bool map_tile(const int* offsets, int rt,
                                         int& e, int& seg, int& cnt, int& row0) {
  for (int i = 0; i < N_EXPERTS; ++i) {
    int c = offsets[i + 1] - offsets[i];
    int tiles = (c + 15) >> 4;
    if (rt < tiles) { e = i; seg = offsets[i]; cnt = c; row0 = rt << 4; return true; }
    rt -= tiles;
  }
  return false;
}

// load B fragment (16 K-contiguous fp32 per lane -> bf16)
__device__ __forceinline__ v16bf load_b_frag(const float* p) {
  const float4* pb = (const float4*)p;
  v16bf b;
#pragma unroll
  for (int j = 0; j < 4; ++j) {
    float4 v = pb[j];
    b[4 * j + 0] = (bf16_t)v.x; b[4 * j + 1] = (bf16_t)v.y;
    b[4 * j + 2] = (bf16_t)v.z; b[4 * j + 3] = (bf16_t)v.w;
  }
  return b;
}

__device__ __forceinline__ float fast_silu(float y) {
  return y * __builtin_amdgcn_rcpf(1.f + __expf(-y));
}

// ---------- GEMM1: h = silu(x_gathered @ w1[e]^T + b1[e]) ----------
// block: 8 waves, tile 16 rows x 256 cols (2 x 16-col tiles per wave)
__global__ __launch_bounds__(256) void gemm1_kernel(
    const float* __restrict__ x, const float* __restrict__ w1,
    const float* __restrict__ b1, const int* __restrict__ offsets,
    const int* __restrict__ perm, bf16_t* __restrict__ hb)
{
  __shared__ bf16_t As[16 * LDA];
  int e, seg, cnt, row0;
  if (!map_tile(offsets, blockIdx.x, e, seg, cnt, row0)) return;
  int tid = threadIdx.x, wave = tid >> 5, lane = tid & 31;

  { // stage 16x1024 gathered A tile, fp32 -> bf16
    int r = tid >> 4, c0 = (tid & 15) * 64;
    int token = (row0 + r < cnt) ? perm[seg + row0 + r] : perm[seg];
    const float4* src = (const float4*)(x + (size_t)token * D_MODEL + c0);
    bf16_t* dst = &As[r * LDA + c0];
#pragma unroll
    for (int i = 0; i < 16; ++i) {
      float4 v = src[i];
      dst[4 * i + 0] = (bf16_t)v.x; dst[4 * i + 1] = (bf16_t)v.y;
      dst[4 * i + 2] = (bf16_t)v.z; dst[4 * i + 3] = (bf16_t)v.w;
    }
  }
  __syncthreads();

  int n0 = blockIdx.y * 256 + wave * 32 + (lane & 15);
  int n1 = n0 + 16;
  int arow = lane & 15;
  int kbA = (lane >> 4) << 4;   // A lane-half: +16 K
  int kbB = (lane >> 4) << 4;   // B lane-half: +16 K
  const float* wrow0 = w1 + (size_t)e * FFN_H * D_MODEL + (size_t)n0 * D_MODEL;
  const float* wrow1 = w1 + (size_t)e * FFN_H * D_MODEL + (size_t)n1 * D_MODEL;

  v8f acc0, acc1;
#pragma unroll
  for (int i = 0; i < 8; ++i) { acc0[i] = 0.f; acc1[i] = 0.f; }

  for (int k0 = 0; k0 < D_MODEL; k0 += 32) {
    v16bf a;
    const bf16_t* pa = &As[arow * LDA + k0 + kbA];
#pragma unroll
    for (int i = 0; i < 8; ++i) { a[i] = pa[i]; a[8 + i] = pa[8 + i]; }
    v16bf b0 = load_b_frag(wrow0 + k0 + kbB);
    v16bf b1f = load_b_frag(wrow1 + k0 + kbB);
    acc0 = __builtin_amdgcn_wmma_f32_16x16x32_bf16(false, a, false, b0,
                                                   (short)0, acc0, false, false);
    acc1 = __builtin_amdgcn_wmma_f32_16x16x32_bf16(false, a, false, b1f,
                                                   (short)0, acc1, false, false);
  }
  float bias0 = b1[e * FFN_H + n0];
  float bias1 = b1[e * FFN_H + n1];
#pragma unroll
  for (int vv = 0; vv < 8; ++vv) {
    int m = vv + ((lane >> 4) << 3);
    if (row0 + m < cnt) {
      size_t rbase = (size_t)(seg + row0 + m) * FFN_H;
      hb[rbase + n0] = (bf16_t)fast_silu(acc0[vv] + bias0);
      hb[rbase + n1] = (bf16_t)fast_silu(acc1[vv] + bias1);
    }
  }
}

// ---------- GEMM2: y = h @ w2[e]^T + b2[e] ----------
__global__ __launch_bounds__(256) void gemm2_kernel(
    const bf16_t* __restrict__ hb, const float* __restrict__ w2,
    const float* __restrict__ b2, const int* __restrict__ offsets,
    float* __restrict__ yb)
{
  __shared__ bf16_t As[16 * LDA];
  int e, seg, cnt, row0;
  if (!map_tile(offsets, blockIdx.x, e, seg, cnt, row0)) return;
  int tid = threadIdx.x, wave = tid >> 5, lane = tid & 31;

  int n0 = blockIdx.y * 256 + wave * 32 + (lane & 15);
  int n1 = n0 + 16;
  int arow = lane & 15;
  int kbA = (lane >> 4) << 4;
  int kbB = (lane >> 4) << 4;
  const float* wrow0 = w2 + (size_t)e * D_MODEL * FFN_H + (size_t)n0 * FFN_H;
  const float* wrow1 = w2 + (size_t)e * D_MODEL * FFN_H + (size_t)n1 * FFN_H;

  v8f acc0, acc1;
#pragma unroll
  for (int i = 0; i < 8; ++i) { acc0[i] = 0.f; acc1[i] = 0.f; }

  for (int kc = 0; kc < FFN_H; kc += 1024) {
    __syncthreads();
    { // stage 16x1024 bf16 chunk of gathered h rows
      int r = tid >> 4, c0 = (tid & 15) * 64;
      int ar = (row0 + r < cnt) ? (seg + row0 + r) : seg;
      const uint4* src = (const uint4*)(hb + (size_t)ar * FFN_H + kc + c0);
      uint4* dst = (uint4*)&As[r * LDA + c0];
#pragma unroll
      for (int i = 0; i < 8; ++i) dst[i] = src[i];
    }
    __syncthreads();
    for (int k0 = 0; k0 < 1024; k0 += 32) {
      v16bf a;
      const bf16_t* pa = &As[arow * LDA + k0 + kbA];
#pragma unroll
      for (int i = 0; i < 8; ++i) { a[i] = pa[i]; a[8 + i] = pa[8 + i]; }
      v16bf b0 = load_b_frag(wrow0 + kc + k0 + kbB);
      v16bf b1f = load_b_frag(wrow1 + kc + k0 + kbB);
      acc0 = __builtin_amdgcn_wmma_f32_16x16x32_bf16(false, a, false, b0,
                                                     (short)0, acc0, false, false);
      acc1 = __builtin_amdgcn_wmma_f32_16x16x32_bf16(false, a, false, b1f,
                                                     (short)0, acc1, false, false);
    }
  }
  float bias0 = b2[e * D_MODEL + n0];
  float bias1 = b2[e * D_MODEL + n1];
#pragma unroll
  for (int vv = 0; vv < 8; ++vv) {
    int m = vv + ((lane >> 4) << 3);
    if (row0 + m < cnt) {
      size_t rbase = (size_t)(seg + row0 + m) * D_MODEL;
      yb[rbase + n0] = acc0[vv] + bias0;
      yb[rbase + n1] = acc1[vv] + bias1;
    }
  }
}

// ---------- combine: out[t] = w0*y[a0] + w1*y[a1] (deterministic) ----------
__global__ __launch_bounds__(256) void combine_kernel(
    const float* __restrict__ yb, const int* __restrict__ assign_of,
    const float* __restrict__ topk_w, float* __restrict__ out, int N)
{
  int idx = blockIdx.x * 256 + threadIdx.x;
  int t = idx >> 8;
  if (t >= N) return;
  int c = (idx & 255) << 2;
  int a0 = assign_of[2 * t], a1 = assign_of[2 * t + 1];
  float w0 = topk_w[2 * t], w1v = topk_w[2 * t + 1];
  float4 y0 = *(const float4*)(yb + (size_t)a0 * D_MODEL + c);
  float4 y1 = *(const float4*)(yb + (size_t)a1 * D_MODEL + c);
  float4 o;
  o.x = w0 * y0.x + w1v * y1.x;  o.y = w0 * y0.y + w1v * y1.y;
  o.z = w0 * y0.z + w1v * y1.z;  o.w = w0 * y0.w + w1v * y1.w;
  *(float4*)(out + (size_t)t * D_MODEL + c) = o;
}

// ---------- aux loss: fixed-order reduction ----------
__global__ __launch_bounds__(256) void aux_kernel(
    const float* __restrict__ probs, const int* __restrict__ counts,
    int N, float* __restrict__ out_aux)
{
  __shared__ float sh[N_EXPERTS * 256];
  float loc[N_EXPERTS];
#pragma unroll
  for (int e = 0; e < N_EXPERTS; ++e) loc[e] = 0.f;
  for (int t = threadIdx.x; t < N; t += 256)
#pragma unroll
    for (int e = 0; e < N_EXPERTS; ++e) loc[e] += probs[t * N_EXPERTS + e];
#pragma unroll
  for (int e = 0; e < N_EXPERTS; ++e) sh[e * 256 + threadIdx.x] = loc[e];
  __syncthreads();
  for (int s = 128; s > 0; s >>= 1) {
    if ((int)threadIdx.x < s)
#pragma unroll
      for (int e = 0; e < N_EXPERTS; ++e)
        sh[e * 256 + threadIdx.x] += sh[e * 256 + threadIdx.x + s];
    __syncthreads();
  }
  if (threadIdx.x == 0) {
    float invNK = 1.f / (float)(2 * N), invN = 1.f / (float)N, aux = 0.f;
#pragma unroll
    for (int e = 0; e < N_EXPERTS; ++e)
      aux += ((float)counts[e] * invNK) * (sh[e * 256] * invN);
    out_aux[0] = (float)N_EXPERTS * aux;
  }
}

extern "C" void kernel_launch(void* const* d_in, const int* in_sizes, int n_in,
                              void* d_out, int out_size, void* d_ws, size_t ws_size,
                              hipStream_t stream) {
  (void)n_in; (void)out_size; (void)ws_size;
  const float* x  = (const float*)d_in[0];
  const float* rw = (const float*)d_in[1];
  const float* w1 = (const float*)d_in[2];
  const float* b1 = (const float*)d_in[3];
  const float* w2 = (const float*)d_in[4];
  const float* b2 = (const float*)d_in[5];
  float* out = (float*)d_out;

  int N = in_sizes[0] / D_MODEL;       // B*T tokens
  if (N <= 0) return;
  int twoN = 2 * N;

  // ----- workspace partition -----
  char* wsb = (char*)d_ws;
  int* hdr = (int*)wsb;                                     // 32 ints
  size_t off = 256;
  int*    topk_idx  = (int*)(wsb + off);  off += (size_t)twoN * 4;
  float*  topk_w    = (float*)(wsb + off); off += (size_t)twoN * 4;
  int*    perm      = (int*)(wsb + off);  off += (size_t)twoN * 4;
  int*    assign_of = (int*)(wsb + off);  off += (size_t)twoN * 4;
  float*  probs     = (float*)(wsb + off); off += (size_t)N * N_EXPERTS * 4;
  off = (off + 255) & ~(size_t)255;
  bf16_t* hb        = (bf16_t*)(wsb + off); off += (size_t)twoN * FFN_H * 2;
  off = (off + 255) & ~(size_t)255;
  float*  yb        = (float*)(wsb + off); off += (size_t)twoN * D_MODEL * 4;

  int* offsets = hdr + 16;
  unsigned maxRowTiles = (unsigned)((twoN + 15) / 16 + N_EXPERTS);

  zero_header_kernel<<<1, 64, 0, stream>>>(hdr);
  router_kernel<<<(unsigned)((N + 7) / 8), 256, 0, stream>>>(
      x, rw, N, hdr, topk_idx, topk_w, probs);
  scan_kernel<<<1, 32, 0, stream>>>(hdr);
  scatter_kernel<<<(unsigned)((twoN + 255) / 256), 256, 0, stream>>>(
      topk_idx, hdr, perm, assign_of, twoN);

  dim3 g1(maxRowTiles, FFN_H / 256);
  gemm1_kernel<<<g1, 256, 0, stream>>>(x, w1, b1, offsets, perm, hb);

  dim3 g2(maxRowTiles, D_MODEL / 256);
  gemm2_kernel<<<g2, 256, 0, stream>>>(hb, w2, b2, offsets, yb);

  combine_kernel<<<(unsigned)N, 256, 0, stream>>>(yb, assign_of, topk_w, out, N);
  aux_kernel<<<1, 256, 0, stream>>>(probs, hdr, N, out + (size_t)N * D_MODEL);
}
